// MultiHeadAttention_25606595019417
// MI455X (gfx1250) — matrix operand order repaired and checked by reading, not compiled, not run on hardware
//
#include <hip/hip_runtime.h>

// Problem constants (match reference)
#define Bq  2
#define Sq  2048
#define Dq  768
#define Hq  12
#define DEP 64
#define Mq  (Bq * Sq)   // 4096 rows in the projection GEMMs

typedef __bf16 bf16x8  __attribute__((ext_vector_type(8)));
typedef __bf16 bf16x16 __attribute__((ext_vector_type(16)));
typedef float  f32x8   __attribute__((ext_vector_type(8)));

__device__ __forceinline__ bf16x16 cat16(bf16x8 lo, bf16x8 hi) {
  return __builtin_shufflevector(lo, hi, 0,1,2,3,4,5,6,7,8,9,10,11,12,13,14,15);
}

__device__ __forceinline__ f32x8 wmma_bf16(bf16x16 a, bf16x16 b, f32x8 c) {
  // 8 args: (neg_a, A, neg_b, B, c_mod, C, reuse_a, reuse_b)
  return __builtin_amdgcn_wmma_f32_16x16x32_bf16(false, a, false, b, (short)0, c, false, false);
}

// A-fragment (16x32 bf16). Lane holds row M = lane%16; half = lane/16:
//   elems 0..7  -> K = kb + 8*half + e        (16B contiguous)
//   elems 8..15 -> K = kb + 16 + 8*half + e-8 (16B contiguous)
__device__ __forceinline__ bf16x16 load_a_frag(const __bf16* row, int kb, int half) {
  bf16x8 lo = *(const bf16x8*)(row + kb + 8 * half);
  bf16x8 hi = *(const bf16x8*)(row + kb + 16 + 8 * half);
  return cat16(lo, hi);
}

// B-fragment (32x16 bf16) read from an [N][K] ("transposed") buffer.
// Lane holds column N (caller passes that row pointer); half = lane/16:
//   elems 0..7  -> K = kb + 8*half + e        (16B contiguous)
//   elems 8..15 -> K = kb + 16 + 8*half + e-8 (16B contiguous)
__device__ __forceinline__ bf16x16 load_b_frag(const __bf16* row, int kb, int half) {
  bf16x8 lo = *(const bf16x8*)(row + kb + 8 * half);
  bf16x8 hi = *(const bf16x8*)(row + kb + 16 + 8 * half);
  return cat16(lo, hi);
}

// ---- CDNA5 async global->LDS copy (ASYNCcnt-tracked, ISA 08_async_tensor) ----
// LDS byte address = low 32 bits of the generic pointer (aperture mapping).
__device__ __forceinline__ unsigned lds_off(const void* p) {
  return (unsigned)(size_t)p;
}
__device__ __forceinline__ void async_ld_b128(unsigned lds_byte, const void* gaddr) {
  asm volatile("global_load_async_to_lds_b128 %0, %1, off"
               :: "v"(lds_byte), "v"(gaddr) : "memory");
}
__device__ __forceinline__ void wait_async4() {   // allow 1 stage (4 ops) in flight
  asm volatile("s_wait_asynccnt 0x4" ::: "memory");
}
__device__ __forceinline__ void wait_async0() {
  asm volatile("s_wait_asynccnt 0x0" ::: "memory");
}

// ---------------------------------------------------------------------------
// fp32 -> bf16 elementwise
__global__ void f32_to_bf16_kernel(const float* __restrict__ in,
                                   __bf16* __restrict__ out, int n) {
  int i = blockIdx.x * blockDim.x + threadIdx.x;
  if (i < n) out[i] = (__bf16)in[i];
}

// W [K x N] fp32 -> Wt [N x K] bf16 (so B-fragments are contiguous loads)
__global__ void transpose_w_kernel(const float* __restrict__ W,
                                   __bf16* __restrict__ Wt) {
  int i = blockIdx.x * blockDim.x + threadIdx.x;   // i = n*Dq + k
  if (i < Dq * Dq) {
    int n = i / Dq, k = i % Dq;
    Wt[i] = (__bf16)W[(size_t)k * Dq + n];
  }
}

// ---------------------------------------------------------------------------
// WMMA GEMM: one wave computes a 32(M) x 64(N) tile: 8 WMMAs per k-step.
// Wave->tile map: the 4 waves of a block share the SAME B strip (nt) so the
// WGP-cache absorbs the dominant (re-read 128x) weight stream; mt varies.
// MODE 0: fp32 row-major out.   MODE 1: bf16 [B,H,S,64].   MODE 2: bf16 [B,H,64,S].
template <int MODE>
__global__ __launch_bounds__(128)
void gemm_bf16_kernel(const __bf16* __restrict__ A,    // [M x 768] row-major bf16
                      const __bf16* __restrict__ Bt,   // [768 x 768] stored [n][k]
                      const float* __restrict__ bias,  // [768]
                      void* __restrict__ out) {
  const int MT = Mq / 32;  // 128 row strips
  int wg = blockIdx.x * (blockDim.x >> 5) + (threadIdx.x >> 5);
  int mt = wg % MT, nt = wg / MT;
  int lane = threadIdx.x & 31, lm = lane & 15, lh = lane >> 4;

  const __bf16* arow0 = A + (size_t)(mt * 32 + lm) * Dq;
  const __bf16* arow1 = arow0 + (size_t)16 * Dq;
  const __bf16* brow0 = Bt + (size_t)(nt * 64 + lm) * Dq;
  const __bf16* brow1 = brow0 + (size_t)16 * Dq;
  const __bf16* brow2 = brow0 + (size_t)32 * Dq;
  const __bf16* brow3 = brow0 + (size_t)48 * Dq;

  f32x8 acc[2][4] = {};
#pragma unroll 2
  for (int kb = 0; kb < Dq; kb += 32) {
    __builtin_prefetch(arow0 + kb + 192, 0, 1);   // ~6 k-steps ahead
    __builtin_prefetch(brow0 + kb + 192, 0, 1);
    bf16x16 a0 = load_a_frag(arow0, kb, lh);
    bf16x16 a1 = load_a_frag(arow1, kb, lh);
    bf16x16 b0 = load_b_frag(brow0, kb, lh);
    bf16x16 b1 = load_b_frag(brow1, kb, lh);
    bf16x16 b2 = load_b_frag(brow2, kb, lh);
    bf16x16 b3 = load_b_frag(brow3, kb, lh);
    acc[0][0] = wmma_bf16(a0, b0, acc[0][0]);
    acc[0][1] = wmma_bf16(a0, b1, acc[0][1]);
    acc[0][2] = wmma_bf16(a0, b2, acc[0][2]);
    acc[0][3] = wmma_bf16(a0, b3, acc[0][3]);
    acc[1][0] = wmma_bf16(a1, b0, acc[1][0]);
    acc[1][1] = wmma_bf16(a1, b1, acc[1][1]);
    acc[1][2] = wmma_bf16(a1, b2, acc[1][2]);
    acc[1][3] = wmma_bf16(a1, b3, acc[1][3]);
  }

#pragma unroll
  for (int j = 0; j < 4; ++j) {
    float bv = bias[nt * 64 + j * 16 + lm];
#pragma unroll
    for (int i = 0; i < 2; ++i) {
#pragma unroll
      for (int r = 0; r < 8; ++r) {
        int mg = mt * 32 + i * 16 + r + 8 * lh;  // C: M = r + 8*(lane/16)
        int ng = nt * 64 + j * 16 + lm;          // C: N = lane%16
        float vv = acc[i][j][r] + bv;
        if (MODE == 0) {
          ((float*)out)[(size_t)mg * Dq + ng] = vv;
        } else {
          int bb = mg >> 11, ss = mg & (Sq - 1);
          int hh = ng >> 6,  dd = ng & 63;
          if (MODE == 1)
            ((__bf16*)out)[(((size_t)(bb * Hq + hh) * Sq) + ss) * DEP + dd] = (__bf16)vv;
          else
            ((__bf16*)out)[(((size_t)(bb * Hq + hh) * DEP) + dd) * Sq + ss] = (__bf16)vv;
        }
      }
    }
  }
}

// ---------------------------------------------------------------------------
// Causal flash attention (transposed form) with block-cooperative, double-
// buffered async staging of K/V tiles into LDS.
//   S^T = K * Q^T   (lane owns a query column; stats are per-lane scalars)
//   O^T += V^T * P^T
// Grid: (B*H)*(S/64) blocks, 128 threads = 4 waves; 16 q-rows per wave.
__global__ __launch_bounds__(128)
void flash_attn_kernel(const __bf16* __restrict__ Q,   // [B*H, S, 64]
                       const __bf16* __restrict__ K,   // [B*H, S, 64]
                       const __bf16* __restrict__ Vt,  // [B*H, 64, S]
                       __bf16* __restrict__ O) {       // [B, S, 768] (head-concat)
  __shared__ __bf16 kbuf[2][32 * 64];   // [key][d]  4KB each, double-buffered
  __shared__ __bf16 vbuf[2][64 * 32];   // [d][key]  4KB each
  __shared__ __bf16 plds[4][16 * 32];   // per-wave P^T tile, stored [q][k]

  const int QT = Sq / 64;  // 32 q-tiles per head
  int bh = blockIdx.x / QT;
  int qt = blockIdx.x % QT;
  int tid  = threadIdx.x;
  // Wave id is wave-uniform: pin it to an SGPR so causal-range guards compile
  // to scalar branches (no EXEC masking around WMMA).
  int wave = __builtin_amdgcn_readfirstlane(tid >> 5);
  int lane = tid & 31, lm = lane & 15, lh = lane >> 4;
  int qb_blk = qt * 64;
  int qbase  = qb_blk + wave * 16;          // scalar

  const __bf16* Qb = Q  + (size_t)bh * Sq * DEP;
  const __bf16* Kb = K  + (size_t)bh * Sq * DEP;
  const __bf16* Vb = Vt + (size_t)bh * DEP * Sq;
  __bf16* pw = plds[wave];

  // Q as B-operand (Q^T): lane holds query column q = qbase + lm
  const __bf16* qrow = Qb + (size_t)(qbase + lm) * DEP;
  bf16x16 qb0 = load_b_frag(qrow, 0,  lh);   // d = 0..31
  bf16x16 qb1 = load_b_frag(qrow, 32, lh);   // d = 32..63

  float m = -1e30f, l = 0.f;               // per-lane scalars (query = qbase+lm)
  f32x8 o0 = {}, o1 = {}, o2 = {}, o3 = {};
  const int qg = qbase + lm;

  // Staging: per stage each wave issues 4 async b128 copies (ASYNCcnt +4).
  // K tile (32x64 bf16 = 4KB) is contiguous in global; V tile is 64 rows x 64B.
  unsigned koff0 = lds_off(&kbuf[0][0]), koff1 = lds_off(&kbuf[1][0]);
  unsigned voff0 = lds_off(&vbuf[0][0]), voff1 = lds_off(&vbuf[1][0]);
  int c0 = tid, c1 = tid + 128;            // two 16B chunks per thread per tile
  auto stage = [&](int buf, int k0) {
    unsigned ko = buf ? koff1 : koff0;
    unsigned vo = buf ? voff1 : voff0;
    const __bf16* kg = Kb + (size_t)k0 * DEP;
    async_ld_b128(ko + (unsigned)c0 * 16, kg + c0 * 8);
    async_ld_b128(ko + (unsigned)c1 * 16, kg + c1 * 8);
    async_ld_b128(vo + (unsigned)c0 * 16, Vb + (size_t)(c0 >> 2) * Sq + k0 + (c0 & 3) * 8);
    async_ld_b128(vo + (unsigned)c1 * 16, Vb + (size_t)(c1 >> 2) * Sq + k0 + (c1 & 3) * 8);
  };

  int nkb_w   = (qbase + 47) >> 5;         // scalar: this wave's causal blocks
  int nfull_w = qbase >> 5;                // scalar: blocks needing no masking
  int nkb_blk = (qb_blk + 95) >> 5;        // block-wide trip count (last wave)

  stage(0, 0);
  for (int kb = 0; kb < nkb_blk; ++kb) {
    int cur = kb & 1;
    if (kb + 1 < nkb_blk) { stage(cur ^ 1, (kb + 1) * 32); wait_async4(); }
    else                  { wait_async0(); }
    __syncthreads();                        // all waves' tile 'cur' staged

    if (kb < nkb_w) {                       // scalar branch: EXEC untouched
      const __bf16* kl = kbuf[cur];         // [32][64]
      const __bf16* vl = vbuf[cur];         // [64][32]
      int k0 = kb * 32;

      // ---- S^T = K * Q^T : two 16(key) x 16(query) tiles ----
      f32x8 s0 = {}, s1 = {};
      s0 = wmma_bf16(load_a_frag(kl + (size_t)lm * DEP, 0,  lh), qb0, s0);
      s0 = wmma_bf16(load_a_frag(kl + (size_t)lm * DEP, 32, lh), qb1, s0);
      s1 = wmma_bf16(load_a_frag(kl + (size_t)(16 + lm) * DEP, 0,  lh), qb0, s1);
      s1 = wmma_bf16(load_a_frag(kl + (size_t)(16 + lm) * DEP, 32, lh), qb1, s1);

      // ---- scale (+ mask only on the single diagonal block) ----
      float pm = -1e30f;
      if (kb < nfull_w) {                   // scalar branch: unmasked fast path
#pragma unroll
        for (int r = 0; r < 8; ++r) {
          float a = s0[r] * 0.125f;
          float b = s1[r] * 0.125f;
          s0[r] = a; s1[r] = b;
          pm = fmaxf(pm, fmaxf(a, b));
        }
      } else {                              // diagonal block
#pragma unroll
        for (int r = 0; r < 8; ++r) {
          int kg = k0 + r + 8 * lh;         // S^T: M = key = r + 8*half
          float a = s0[r] * 0.125f + (kg      > qg ? -1e9f : 0.f);
          float b = s1[r] * 0.125f + (kg + 16 > qg ? -1e9f : 0.f);
          s0[r] = a; s1[r] = b;
          pm = fmaxf(pm, fmaxf(a, b));
        }
      }
      pm = fmaxf(pm, __shfl_xor(pm, 16, 32));   // combine key halves

      float mn   = fmaxf(m, pm);
      float corr = __expf(m - mn);
      float ps = 0.f;
      bf16x8 pk0, pk1;
#pragma unroll
      for (int r = 0; r < 8; ++r) {
        float e0 = __expf(s0[r] - mn);
        float e1 = __expf(s1[r] - mn);
        ps += e0 + e1;
        pk0[r] = (__bf16)e0;
        pk1[r] = (__bf16)e1;
      }
      ps += __shfl_xor(ps, 16, 32);
      l = l * corr + ps;
      m = mn;
      o0 = o0 * corr; o1 = o1 * corr; o2 = o2 * corr; o3 = o3 * corr;

      // ---- P^T -> LDS [q][k] (two packed 16B stores per lane) ----
      *(bf16x8*)(pw + lm * 32 + 8 * lh)      = pk0;   // keys  r+8*lh
      *(bf16x8*)(pw + lm * 32 + 16 + 8 * lh) = pk1;   // keys 16+r+8*lh
      bf16x16 pb = load_b_frag(pw + lm * 32, 0, lh);  // B-frag: K = 8*half + e (e<8), 16 + 8*half + e-8 (e>=8)

      // ---- O^T += V^T * P^T : 4 d-tiles ----
      o0 = wmma_bf16(load_a_frag(vl + (size_t)(0 * 16 + lm) * 32, 0, lh), pb, o0);
      o1 = wmma_bf16(load_a_frag(vl + (size_t)(1 * 16 + lm) * 32, 0, lh), pb, o1);
      o2 = wmma_bf16(load_a_frag(vl + (size_t)(2 * 16 + lm) * 32, 0, lh), pb, o2);
      o3 = wmma_bf16(load_a_frag(vl + (size_t)(3 * 16 + lm) * 32, 0, lh), pb, o3);
    }
    __syncthreads();                        // tile 'cur' free for restaging
  }

  // ---- normalize and store: O^T element (d=t*16+r+8*lh, q=qbase+lm)
  //      -> out[b][s=q][h*64+d]; r runs contiguously in d => packed 16B stores.
  float inv = 1.0f / l;
  int bb = bh / Hq, hh = bh % Hq;
  __bf16* orow = O + (size_t)(bb * Sq + qbase + lm) * Dq + hh * DEP + 8 * lh;
  f32x8 ot[4] = {o0, o1, o2, o3};
#pragma unroll
  for (int t = 0; t < 4; ++t) {
    bf16x8 pack;
#pragma unroll
    for (int r = 0; r < 8; ++r) pack[r] = (__bf16)(ot[t][r] * inv);
    *(bf16x8*)(orow + t * 16) = pack;
  }
}

// ---------------------------------------------------------------------------
extern "C" void kernel_launch(void* const* d_in, const int* in_sizes, int n_in,
                              void* d_out, int out_size, void* d_ws, size_t ws_size,
                              hipStream_t stream) {
  (void)in_sizes; (void)n_in; (void)out_size; (void)ws_size;
  const float* q  = (const float*)d_in[0];
  const float* k  = (const float*)d_in[1];
  const float* v  = (const float*)d_in[2];
  /* d_in[3] = mask: causal triu, applied analytically in-kernel */
  const float* Wq = (const float*)d_in[4];  const float* bq = (const float*)d_in[5];
  const float* Wk = (const float*)d_in[6];  const float* bk = (const float*)d_in[7];
  const float* Wv = (const float*)d_in[8];  const float* bv = (const float*)d_in[9];
  const float* Wo = (const float*)d_in[10]; const float* bo = (const float*)d_in[11];

  const size_t SZ1 = (size_t)Mq * Dq;  // 3,145,728 elems
  const size_t SZW = (size_t)Dq * Dq;  //   589,824 elems
  __bf16* xq  = (__bf16*)d_ws;         // activations in bf16
  __bf16* xk  = xq  + SZ1;
  __bf16* xv  = xk  + SZ1;
  __bf16* wqT = xv  + SZ1;             // weights, transposed bf16 [n][k]
  __bf16* wkT = wqT + SZW;
  __bf16* wvT = wkT + SZW;
  __bf16* woT = wvT + SZW;
  __bf16* Qh  = woT + SZW;             // projected Q  [B,H,S,64]
  __bf16* Kh  = Qh  + SZ1;             // projected K  [B,H,S,64]
  __bf16* Vt  = Kh  + SZ1;             // projected V  [B,H,64,S]
  __bf16* At  = Vt  + SZ1;             // attn output  [B,S,768]

  f32_to_bf16_kernel<<<(Mq * Dq) / 256, 256, 0, stream>>>(q, xq, Mq * Dq);
  f32_to_bf16_kernel<<<(Mq * Dq) / 256, 256, 0, stream>>>(k, xk, Mq * Dq);
  f32_to_bf16_kernel<<<(Mq * Dq) / 256, 256, 0, stream>>>(v, xv, Mq * Dq);
  transpose_w_kernel<<<(Dq * Dq) / 256, 256, 0, stream>>>(Wq, wqT);
  transpose_w_kernel<<<(Dq * Dq) / 256, 256, 0, stream>>>(Wk, wkT);
  transpose_w_kernel<<<(Dq * Dq) / 256, 256, 0, stream>>>(Wv, wvT);
  transpose_w_kernel<<<(Dq * Dq) / 256, 256, 0, stream>>>(Wo, woT);

  int gemm_blocks = (Mq / 32) * (Dq / 64) / 4;  // 384 blocks, 4 waves each
  gemm_bf16_kernel<1><<<gemm_blocks, 128, 0, stream>>>(xq, wqT, bq, Qh);
  gemm_bf16_kernel<1><<<gemm_blocks, 128, 0, stream>>>(xk, wkT, bk, Kh);
  gemm_bf16_kernel<2><<<gemm_blocks, 128, 0, stream>>>(xv, wvT, bv, Vt);

  flash_attn_kernel<<<Bq * Hq * (Sq / 64), 128, 0, stream>>>(Qh, Kh, Vt, At);

  gemm_bf16_kernel<0><<<gemm_blocks, 128, 0, stream>>>(At, woT, bo, d_out);
}
